// SecureLightGCN_24524263260330
// MI455X (gfx1250) — matrix operand order repaired and checked by reading, not compiled, not run on hardware
//
#include <hip/hip_runtime.h>
#include <hip/hip_bf16.h>
#include <math.h>

typedef __attribute__((ext_vector_type(16))) _Float16 v16h;
typedef __attribute__((ext_vector_type(8)))  float    v8f;

#define EMB 64
// ws layout (float indices). Needs ~(1536+512)*4 bytes = 8KB of d_ws.
#define W_OFF      0     // w[0..127] f32
#define SCONST_OFF 128   // s_const
#define GMAX_OFF   129   // global max
#define INVSUM_OFF 130   // 1/sum
#define WH_BYTE_OFF 640  // 64 f16 item-part weights (occupies wsf[160..192))
#define MAXP_OFF   512   // per-block max partials (<=1024 blocks)
#define SUMP_OFF   1536  // per-block sum partials

// ---------------------------------------------------------------------------
// K0: fold the two linear layers: w[j] = sum_k W2[0,k]*W1[k,j]  (W1 is [64,128])
//     s_const = user_emb . w[0:64] + (b1.W2 + b2)
// ---------------------------------------------------------------------------
__global__ void k_prep(const float* __restrict__ user_table,
                       const float* __restrict__ W1,
                       const float* __restrict__ b1,
                       const float* __restrict__ W2,
                       const float* __restrict__ b2,
                       const int*   __restrict__ uidx,
                       float* __restrict__ wsf,
                       _Float16* __restrict__ wh) {
  __shared__ float lw[128];
  int j = threadIdx.x;                    // 128 threads
  float wj = 0.f;
  #pragma unroll 4
  for (int k = 0; k < EMB; ++k) wj = fmaf(W2[k], W1[k * 128 + j], wj);
  wsf[W_OFF + j] = wj;
  lw[j] = wj;
  if (j >= 64) wh[j - 64] = (_Float16)wj; // item-part weights as f16 (B matrix)
  __syncthreads();
  if (j == 0) {
    const float* ue = user_table + (long long)uidx[0] * EMB;
    float s = 0.f;
    for (int k = 0; k < EMB; ++k) s = fmaf(ue[k], lw[k], s);
    float cb = b2[0];
    for (int k = 0; k < EMB; ++k) cb = fmaf(W2[k], b1[k], cb);
    wsf[SCONST_OFF] = s + cb;
  }
}

__device__ __forceinline__ float sel8(const v8f& c, int j) {
  float y = c[0];
  y = (j == 1) ? (float)c[1] : y;
  y = (j == 2) ? (float)c[2] : y;
  y = (j == 3) ? (float)c[3] : y;
  y = (j == 4) ? (float)c[4] : y;
  y = (j == 5) ? (float)c[5] : y;
  y = (j == 6) ? (float)c[6] : y;
  y = (j == 7) ? (float)c[7] : y;
  return y;
}

// ---------------------------------------------------------------------------
// K1: gathered mat-vec via WMMA. One wave = 16 items, K=64 via two
//     v_wmma_f32_16x16x32_f16 ops. A per ISA 16-bit A layout; B = weight
//     vector broadcast across all 16 columns, so D[m,n] = y[m] for every n.
// ---------------------------------------------------------------------------
__global__ __launch_bounds__(256) void k_att(
    const float* __restrict__ item_table,
    const int*   __restrict__ idx,
    const float* __restrict__ wsf,
    const _Float16* __restrict__ wh,
    float* __restrict__ att_out,
    float* __restrict__ maxp, int N) {
  const int lane = threadIdx.x & 31;
  const int gw   = blockIdx.x * (blockDim.x >> 5) + (threadIdx.x >> 5);
  const int base = gw * 16;
  float myv = -INFINITY;

  if (base < N) {                                  // wave-uniform: EXEC stays full
    const int h    = lane >> 4;                    // lane half
    const int m16  = lane & 15;                    // row of A handled by this lane
    int gi = base + m16; if (gi >= N) gi = N - 1;  // clamp (keeps EXEC all-1s)
    const long long row = (long long)idx[gi];
    const float* rp = item_table + row * EMB;
    const int koff  = h * 8;                       // A: lanes>=16 hold K+8
    const float sconst = wsf[SCONST_OFF];

    v8f acc = {};
    #pragma unroll
    for (int cb = 0; cb < EMB; cb += 32) {
      // A layout 16x32 f16: VGPR0-3 -> K = cb+koff+[0,8), VGPR4-7 -> K = cb+16+koff+[0,8)
      float4 q0 = *(const float4*)(rp + cb + koff);
      float4 q1 = *(const float4*)(rp + cb + koff + 4);
      float4 q2 = *(const float4*)(rp + cb + 16 + koff);
      float4 q3 = *(const float4*)(rp + cb + 16 + koff + 4);
      v16h a;
      a[0]=(_Float16)q0.x; a[1]=(_Float16)q0.y; a[2]=(_Float16)q0.z; a[3]=(_Float16)q0.w;
      a[4]=(_Float16)q1.x; a[5]=(_Float16)q1.y; a[6]=(_Float16)q1.z; a[7]=(_Float16)q1.w;
      a[8]=(_Float16)q2.x; a[9]=(_Float16)q2.y; a[10]=(_Float16)q2.z; a[11]=(_Float16)q2.w;
      a[12]=(_Float16)q3.x; a[13]=(_Float16)q3.y; a[14]=(_Float16)q3.z; a[15]=(_Float16)q3.w;
      // B layout 32x16 f16 (broadcast over N): same K order as A:
      // elements 0-7 -> K = cb+koff+[0,8), elements 8-15 -> K = cb+16+koff+[0,8).
      const _Float16* wp = wh + cb + 8 * h;
      v16h b;
      #pragma unroll
      for (int t = 0; t < 16; ++t) b[t] = wp[t < 8 ? t : t + 8];
      acc = __builtin_amdgcn_wmma_f32_16x16x32_f16(
          /*neg_a=*/false, a, /*neg_b=*/false, b,
          /*c_mod=*/(short)0, acc, /*reuse_a=*/false, /*reuse_b=*/false);
    }
    // D: lane<16 holds y[0..7] in acc[0..7]; lane>=16 holds y[8..15].
    const int j = lane & 7;
    float y = sel8(acc, j);
    float t = sconst + y;
    float att = t > 0.f ? t : 0.01f * t;           // leaky_relu(0.01)
    const bool writer = ((lane >> 3) & 1) == 0;    // lanes 0-7 and 16-23
    const int m = j + 8 * h;
    if (writer && (base + m) < N) {
      att_out[base + m] = att;
      myv = att;
    }
  }
  // deterministic block max
  __shared__ float red[256];
  red[threadIdx.x] = myv;
  __syncthreads();
  for (int s = 128; s > 0; s >>= 1) {
    if (threadIdx.x < s)
      red[threadIdx.x] = fmaxf(red[threadIdx.x], red[threadIdx.x + s]);
    __syncthreads();
  }
  if (threadIdx.x == 0) maxp[blockIdx.x] = red[0];
}

__global__ void k_max(const float* __restrict__ maxp, int nb,
                      float* __restrict__ wsf) {
  __shared__ float red[256];
  float m = -INFINITY;
  for (int i = threadIdx.x; i < nb; i += 256) m = fmaxf(m, maxp[i]);
  red[threadIdx.x] = m;
  __syncthreads();
  for (int s = 128; s > 0; s >>= 1) {
    if (threadIdx.x < s)
      red[threadIdx.x] = fmaxf(red[threadIdx.x], red[threadIdx.x + s]);
    __syncthreads();
  }
  if (threadIdx.x == 0) wsf[GMAX_OFF] = red[0];
}

__global__ __launch_bounds__(256) void k_exp(float* __restrict__ att, int N,
                                             const float* __restrict__ wsf,
                                             float* __restrict__ sump) {
  __shared__ float red[256];
  const int i = blockIdx.x * 256 + threadIdx.x;
  const float gmax = wsf[GMAX_OFF];
  float e = 0.f;
  if (i < N) {
    e = expf(att[i] - gmax);
    att[i] = e;
  }
  red[threadIdx.x] = e;
  __syncthreads();
  for (int s = 128; s > 0; s >>= 1) {
    if (threadIdx.x < s) red[threadIdx.x] += red[threadIdx.x + s];
    __syncthreads();
  }
  if (threadIdx.x == 0) sump[blockIdx.x] = red[0];
}

__global__ void k_sum(const float* __restrict__ sump, int nb,
                      float* __restrict__ wsf) {
  __shared__ float red[256];
  float s = 0.f;
  for (int i = threadIdx.x; i < nb; i += 256) s += sump[i];   // fixed order
  red[threadIdx.x] = s;
  __syncthreads();
  for (int st = 128; st > 0; st >>= 1) {
    if (threadIdx.x < st) red[threadIdx.x] += red[threadIdx.x + st];
    __syncthreads();
  }
  if (threadIdx.x == 0) wsf[INVSUM_OFF] = 1.0f / red[0];
}

__global__ __launch_bounds__(256) void k_norm(float* __restrict__ out, int N,
                                              const float* __restrict__ wsf) {
  const int i = blockIdx.x * 256 + threadIdx.x;
  if (i < N) out[i] *= wsf[INVSUM_OFF];
}

extern "C" void kernel_launch(void* const* d_in, const int* in_sizes, int n_in,
                              void* d_out, int out_size, void* d_ws, size_t ws_size,
                              hipStream_t stream) {
  const float* user_table = (const float*)d_in[0];
  const float* item_table = (const float*)d_in[1];
  const float* W1         = (const float*)d_in[2];
  const float* b1         = (const float*)d_in[3];
  const float* W2         = (const float*)d_in[4];
  const float* b2         = (const float*)d_in[5];
  const int*   uidx       = (const int*)d_in[6];
  const int*   idx        = (const int*)d_in[7];
  const int N = in_sizes[7];

  float*    wsf  = (float*)d_ws;
  _Float16* wh   = (_Float16*)((char*)d_ws + WH_BYTE_OFF);
  float*    maxp = wsf + MAXP_OFF;
  float*    sump = wsf + SUMP_OFF;
  float*    out  = (float*)d_out;

  const int numWaves = (N + 15) / 16;
  const int nb1 = (numWaves + 7) / 8;     // 8 waves (256 thr) per block -> 782
  const int nb3 = (N + 255) / 256;        // 391

  k_prep<<<1, 128, 0, stream>>>(user_table, W1, b1, W2, b2, uidx, wsf, wh);
  k_att <<<nb1, 256, 0, stream>>>(item_table, idx, wsf, wh, out, maxp, N);
  k_max <<<1, 256, 0, stream>>>(maxp, nb1, wsf);
  k_exp <<<nb3, 256, 0, stream>>>(out, N, wsf, sump);
  k_sum <<<1, 256, 0, stream>>>(sump, nb3, wsf);
  k_norm<<<nb3, 256, 0, stream>>>(out, N, wsf);
}
